// MultiHeadAttention_28827820490851
// MI455X (gfx1250) — matrix operand order repaired and checked by reading, not compiled, not run on hardware
//
#include <hip/hip_runtime.h>
#include <hip/hip_bf16.h>

#define BATCH 2
#define SEQ   2048
#define DMODEL 1024
#define NHEAD 16
#define HD    64

typedef __attribute__((ext_vector_type(16))) __bf16 v16bf;
typedef __attribute__((ext_vector_type(8)))  float  v8f;

union BF16Frag {
  v16bf b;
  unsigned int d[8];
};

__device__ __forceinline__ unsigned short f32_to_bf16(float f) {
  unsigned int x = __float_as_uint(f);
  unsigned int r = x + 0x7FFFu + ((x >> 16) & 1u);   // round-to-nearest-even
  return (unsigned short)(r >> 16);
}

__device__ __forceinline__ unsigned int pack_bf16(float lo, float hi) {
  return (unsigned int)f32_to_bf16(lo) | ((unsigned int)f32_to_bf16(hi) << 16);
}

// 16-bit A-fragment (16x32): VGPR v holds packed pair starting at this even K.
__device__ __forceinline__ int kpairA(int v, int half) {
  return ((v < 4) ? (2 * v) : (16 + 2 * (v - 4))) + 8 * half;
}
// 16-bit B-fragment (32x16): packed pair starting K.
__device__ __forceinline__ int kpairB(int v, int half) {
  return ((v < 4) ? (2 * v) : (16 + 2 * (v - 4))) + 8 * half;
}

__device__ __forceinline__ v8f wmma_bf16(const BF16Frag& a, const BF16Frag& b, v8f c) {
  return __builtin_amdgcn_wmma_f32_16x16x32_bf16(
      false, a.b, false, b.b, (short)0, c, false, false);
}

// ---- CDNA5 async copy: global -> LDS, 16 bytes per lane (ASYNCcnt) ----
__device__ __forceinline__ void async_copy16(unsigned lds_off, const void* gptr) {
  asm volatile("global_load_async_to_lds_b128 %0, %1, off"
               :: "v"(lds_off), "v"(gptr)
               : "memory");
}
__device__ __forceinline__ void wait_async_le(int n) {
  if (n == 0) asm volatile("s_wait_asynccnt 0x0" ::: "memory");
  else        asm volatile("s_wait_asynccnt 0x2" ::: "memory");
}

// ---- butterfly reductions across the 16-lane half (VALU permlane if present) --
__device__ __forceinline__ float bfly16(float x, int step) {
#if __has_builtin(__builtin_amdgcn_permlane16)
  // nibble selects encoding lane i <- lane (i ^ step) within rows of 16
  unsigned s1, s2;
  switch (step) {
    case 1: s1 = 0x67452301u; s2 = 0xEFCDAB89u; break;
    case 2: s1 = 0x54761032u; s2 = 0xDCFE98BAu; break;
    case 4: s1 = 0x32107654u; s2 = 0xBA98FEDCu; break;
    default: s1 = 0xFEDCBA98u; s2 = 0x76543210u; break; // 8
  }
  unsigned v = __float_as_uint(x);
  return __uint_as_float(__builtin_amdgcn_permlane16(v, v, s1, s2, false, false));
#else
  return __shfl_xor(x, step, 32);
#endif
}
__device__ __forceinline__ float half16_max(float x) {
  x = fmaxf(x, bfly16(x, 1));
  x = fmaxf(x, bfly16(x, 2));
  x = fmaxf(x, bfly16(x, 4));
  x = fmaxf(x, bfly16(x, 8));
  return x;
}
__device__ __forceinline__ float half16_sum(float x) {
  x += bfly16(x, 1);
  x += bfly16(x, 2);
  x += bfly16(x, 4);
  x += bfly16(x, 8);
  return x;
}

// ---------------------------------------------------------------------------
// Kernel 1: fp32 -> bf16 conversion (weights)
// ---------------------------------------------------------------------------
__global__ void cvt_f32_bf16_kernel(const float* __restrict__ src,
                                    unsigned short* __restrict__ dst, int n) {
  int i = blockIdx.x * blockDim.x + threadIdx.x;
  if (i < n) dst[i] = f32_to_bf16(src[i]);
}

// ---------------------------------------------------------------------------
// Kernel 2: tiled GEMM  Y = X[4096x1024] @ W[1024x1024] + bias
// Block: 256 threads (8 waves); block tile 128 rows x 64 cols.
// Each wave: 1 A-fragment reused over 4 B-fragments -> 4 WMMA per k-step.
// IN_IS_BF16: 0 = X fp32 (convert while staging), 1 = X bf16 (async-to-LDS)
// OUT_MODE:   0 = bf16 [b,h,s,hd] (Q,K); 1 = bf16 [b,h,hd,s] (V^T); 2 = fp32 [b,s,d]
// ---------------------------------------------------------------------------
template <int IN_IS_BF16, int OUT_MODE>
__global__ void __launch_bounds__(256)
gemm1024_kernel(const void* __restrict__ Xv,
                const unsigned short* __restrict__ Wbf,
                const float* __restrict__ bias,
                void* __restrict__ outv) {
  __shared__ __align__(16) unsigned short As[128][40];  // [row][k] bf16 (pad)
  __shared__ __align__(16) unsigned short Bs[64][40];   // [n][k]   bf16 (pad)

  const int tid  = threadIdx.x;
  const int w    = tid >> 5;
  const int lane = tid & 31;
  const int half = lane >> 4;
  const int ln   = lane & 15;
  const int n0   = blockIdx.x * 64;
  const int row0 = blockIdx.y * 128;

  v8f acc[4];
  v8f z8 = {};
#pragma unroll
  for (int g = 0; g < 4; ++g) acc[g] = z8;

  for (int kt = 0; kt < DMODEL / 32; ++kt) {
    const int k0 = kt * 32;
    if (kt) __syncthreads();

    // ---- stage A tile: 128 rows x 32 k ----
    if (IN_IS_BF16) {
      const unsigned short* X = (const unsigned short*)Xv;
#pragma unroll
      for (int i = 0; i < 2; ++i) {          // 512 chunks of 16B
        int idx = tid + 256 * i;
        int r = idx >> 2, c16 = idx & 3;
        async_copy16((unsigned)(size_t)&As[r][c16 * 8],
                     X + (size_t)(row0 + r) * DMODEL + k0 + c16 * 8);
      }
    } else {
      const float* X = (const float*)Xv;
#pragma unroll
      for (int i = 0; i < 4; ++i) {          // 1024 float4
        int idx = tid + 256 * i;
        int r = idx >> 3, c4 = idx & 7;
        const float* p = X + (size_t)(row0 + r) * DMODEL + k0 + c4 * 4;
        float x0 = p[0], x1 = p[1], x2 = p[2], x3 = p[3];
        unsigned int* dst = (unsigned int*)&As[r][c4 * 4];
        dst[0] = pack_bf16(x0, x1);
        dst[1] = pack_bf16(x2, x3);
      }
    }
    // ---- stage B tile transposed: Bs[n][k] = W[k0+k][n0+n], 32x64 elems ----
    {
      int kk = tid >> 3, n8 = (tid & 7) * 8;
      uint4 wv = *(const uint4*)(Wbf + (size_t)(k0 + kk) * DMODEL + n0 + n8);
      Bs[n8 + 0][kk] = (unsigned short)(wv.x);
      Bs[n8 + 1][kk] = (unsigned short)(wv.x >> 16);
      Bs[n8 + 2][kk] = (unsigned short)(wv.y);
      Bs[n8 + 3][kk] = (unsigned short)(wv.y >> 16);
      Bs[n8 + 4][kk] = (unsigned short)(wv.z);
      Bs[n8 + 5][kk] = (unsigned short)(wv.z >> 16);
      Bs[n8 + 6][kk] = (unsigned short)(wv.w);
      Bs[n8 + 7][kk] = (unsigned short)(wv.w >> 16);
    }
    if (IN_IS_BF16) wait_async_le(0);
    __syncthreads();

    // ---- fragments + 4 WMMA ----
    BF16Frag a;
    const int m = w * 16 + ln;
#pragma unroll
    for (int v = 0; v < 8; ++v)
      a.d[v] = *(const unsigned int*)&As[m][kpairA(v, half)];
#pragma unroll
    for (int g = 0; g < 4; ++g) {
      BF16Frag bf;
#pragma unroll
      for (int v = 0; v < 8; ++v)
        bf.d[v] = *(const unsigned int*)&Bs[g * 16 + ln][kpairB(v, half)];
      acc[g] = wmma_bf16(a, bf, acc[g]);
    }
  }

  // ---- epilogue: bias + scatter ----
#pragma unroll
  for (int g = 0; g < 4; ++g) {
    const int gn = n0 + g * 16 + ln;
    const float bv = bias[gn];
#pragma unroll
    for (int v = 0; v < 8; ++v) {
      const int gm = row0 + w * 16 + v + 8 * half;
      const float val = acc[g][v] + bv;
      if (OUT_MODE == 2) {
        ((float*)outv)[(size_t)gm * DMODEL + gn] = val;
      } else {
        const int bb = gm >> 11, s = gm & (SEQ - 1);
        const int hh = gn >> 6,  hd = gn & (HD - 1);
        unsigned short* o = (unsigned short*)outv;
        if (OUT_MODE == 0)
          o[((size_t)(bb * NHEAD + hh) * SEQ + s) * HD + hd] = f32_to_bf16(val);
        else
          o[((size_t)(bb * NHEAD + hh) * HD + hd) * SEQ + s] = f32_to_bf16(val);
      }
    }
  }
}

// ---------------------------------------------------------------------------
// Kernel 3: flash attention. Q,K in [b,h,s,hd] bf16; V in [b,h,hd,s] bf16.
// Block = 8 waves sharing one (b,h); each wave owns 16 query rows.
// K/V chunks (32 keys) double-buffered into LDS via async-to-LDS copies.
// Per chunk per wave: 4 WMMA (QK^T) + online softmax + 4 WMMA (P.V).
// ---------------------------------------------------------------------------
__global__ void __launch_bounds__(256)
attn_kernel(const unsigned short* __restrict__ Qb,
            const unsigned short* __restrict__ Kb,
            const unsigned short* __restrict__ Vt,
            unsigned short* __restrict__ Ob) {
  __shared__ __align__(16) unsigned short Ks[2][32][72];   // [buf][key][hd]
  __shared__ __align__(16) unsigned short Vs[2][64][40];   // [buf][hd][kk]
  __shared__ __align__(16) unsigned short Ps[8][16 * 34];  // per-wave P tile

  const int tid  = threadIdx.x;
  const int w    = tid >> 5;
  const int lane = tid & 31;
  const int half = lane >> 4;
  const int ln   = lane & 15;
  const int b    = blockIdx.z;
  const int h    = blockIdx.y;
  const int q0   = (blockIdx.x * 8 + w) * 16;

  const unsigned short* Qh = Qb + (size_t)(b * NHEAD + h) * SEQ * HD;
  const unsigned short* Kh = Kb + (size_t)(b * NHEAD + h) * SEQ * HD;
  const unsigned short* Vh = Vt + (size_t)(b * NHEAD + h) * HD * SEQ;

  // Q fragments: two k-steps covering HD=64
  BF16Frag qa[2];
#pragma unroll
  for (int c0 = 0; c0 < 2; ++c0)
#pragma unroll
    for (int v = 0; v < 8; ++v)
      qa[c0].d[v] = *(const unsigned int*)(Qh + (size_t)(q0 + ln) * HD + c0 * 32 + kpairA(v, half));

  float m_i[8], l_i[8];
  v8f acc[4];
  v8f z8 = {};
#pragma unroll
  for (int v = 0; v < 8; ++v) { m_i[v] = -1e30f; l_i[v] = 0.0f; }
#pragma unroll
  for (int g = 0; g < 4; ++g) acc[g] = z8;

  unsigned short* prow = &Ps[w][0];
  const float SCL = 0.125f * 1.44269504088896340736f;  // 1/sqrt(HD) * log2(e)

  // per-thread async copy slices: K = 256 x 16B chunks, V = 256 x 16B chunks
  const int kKey = tid >> 3, kC16 = tid & 7;   // K: [32 keys][8 x 16B]
  const int vHd  = tid >> 2, vC16 = tid & 3;   // V: [64 hd]  [4 x 16B]

  // prologue: issue chunk 0 into buffer 0
  async_copy16((unsigned)(size_t)&Ks[0][kKey][kC16 * 8],
               Kh + (size_t)kKey * HD + kC16 * 8);
  async_copy16((unsigned)(size_t)&Vs[0][vHd][vC16 * 8],
               Vh + (size_t)vHd * SEQ + vC16 * 8);

  const int nChunks = SEQ / 32;
  for (int kc = 0; kc < nChunks; ++kc) {
    const int buf = kc & 1;
    __syncthreads();  // everyone done reading the buffer we are about to fill
    if (kc + 1 < nChunks) {
      const int kk1 = (kc + 1) * 32;
      async_copy16((unsigned)(size_t)&Ks[buf ^ 1][kKey][kC16 * 8],
                   Kh + (size_t)(kk1 + kKey) * HD + kC16 * 8);
      async_copy16((unsigned)(size_t)&Vs[buf ^ 1][vHd][vC16 * 8],
                   Vh + (size_t)vHd * SEQ + kk1 + vC16 * 8);
      wait_async_le(2);   // retire current buffer's copies, keep next in flight
    } else {
      wait_async_le(0);
    }
    __syncthreads();      // all waves' copies for `buf` complete

    const int kk0 = kc * 32;
    (void)kk0;

    // ---- scores: two 16x16 tiles (keys 0..15 and 16..31 of the chunk) ----
    v8f s0 = z8, s1 = z8;
    BF16Frag kb;
#pragma unroll
    for (int c0 = 0; c0 < 2; ++c0) {
#pragma unroll
      for (int v = 0; v < 8; ++v)
        kb.d[v] = *(const unsigned int*)&Ks[buf][ln][c0 * 32 + kpairB(v, half)];
      s0 = wmma_bf16(qa[c0], kb, s0);
    }
#pragma unroll
    for (int c0 = 0; c0 < 2; ++c0) {
#pragma unroll
      for (int v = 0; v < 8; ++v)
        kb.d[v] = *(const unsigned int*)&Ks[buf][16 + ln][c0 * 32 + kpairB(v, half)];
      s1 = wmma_bf16(qa[c0], kb, s1);
    }

    // ---- online softmax (log2 domain); row m = v + 8*half across 16 lanes --
#pragma unroll
    for (int v = 0; v < 8; ++v) {
      float a0 = s0[v] * SCL;
      float a1 = s1[v] * SCL;
      float mx = half16_max(fmaxf(a0, a1));
      float mnew = fmaxf(m_i[v], mx);
      float sc = exp2f(m_i[v] - mnew);
      float p0 = exp2f(a0 - mnew);
      float p1 = exp2f(a1 - mnew);
      l_i[v] = l_i[v] * sc + (p0 + p1);   // per-lane partial; reduce at end
      m_i[v] = mnew;
#pragma unroll
      for (int g = 0; g < 4; ++g) acc[g][v] *= sc;
      const int m = v + 8 * half;
      prow[m * 34 + ln]      = f32_to_bf16(p0);
      prow[m * 34 + 16 + ln] = f32_to_bf16(p1);
    }

    // ---- reload P as A-fragment (in-wave LDS ordering, no barrier needed) --
    BF16Frag pa;
#pragma unroll
    for (int v = 0; v < 8; ++v)
      pa.d[v] = *(const unsigned int*)&prow[ln * 34 + kpairA(v, half)];

    // ---- acc += P (16x32) @ V (32x64): 4 WMMA over hd groups of 16 ----
#pragma unroll
    for (int g = 0; g < 4; ++g) {
      BF16Frag vb;
#pragma unroll
      for (int v = 0; v < 8; ++v)
        vb.d[v] = *(const unsigned int*)&Vs[buf][g * 16 + ln][kpairB(v, half)];
      acc[g] = wmma_bf16(pa, vb, acc[g]);
    }
  }

  // ---- epilogue: reduce l across the half, normalize, write bf16 ----
#pragma unroll
  for (int v = 0; v < 8; ++v) {
    const float inv = 1.0f / half16_sum(l_i[v]);
    const int m = v + 8 * half;
    const size_t row = (size_t)(b * SEQ + q0 + m) * DMODEL + h * HD;
#pragma unroll
    for (int g = 0; g < 4; ++g)
      Ob[row + g * 16 + ln] = f32_to_bf16(acc[g][v] * inv);
  }
}

// ---------------------------------------------------------------------------
// Host-side launch
// ---------------------------------------------------------------------------
extern "C" void kernel_launch(void* const* d_in, const int* in_sizes, int n_in,
                              void* d_out, int out_size, void* d_ws, size_t ws_size,
                              hipStream_t stream) {
  (void)in_sizes; (void)n_in; (void)out_size; (void)ws_size;

  const float* query = (const float*)d_in[0];
  const float* key   = (const float*)d_in[1];
  const float* value = (const float*)d_in[2];
  const float* Wq    = (const float*)d_in[3];
  const float* bq    = (const float*)d_in[4];
  const float* Wk    = (const float*)d_in[5];
  const float* bk    = (const float*)d_in[6];
  const float* Wv    = (const float*)d_in[7];
  const float* bv    = (const float*)d_in[8];
  const float* Wo    = (const float*)d_in[9];
  const float* bo    = (const float*)d_in[10];

  const size_t WE = (size_t)DMODEL * DMODEL;          // 1M elems per weight
  const size_t AE = (size_t)BATCH * SEQ * DMODEL;     // 4M elems per activation

  unsigned short* wsp = (unsigned short*)d_ws;
  unsigned short* Wqb = wsp + 0 * WE;
  unsigned short* Wkb = wsp + 1 * WE;
  unsigned short* Wvb = wsp + 2 * WE;
  unsigned short* Wob = wsp + 3 * WE;
  unsigned short* Qbf = wsp + 4 * WE;
  unsigned short* Kbf = Qbf + AE;
  unsigned short* Vtb = Kbf + AE;   // V stored transposed [b,h,hd,s]
  unsigned short* Obf = Vtb + AE;

  // 1) weights fp32 -> bf16
  const int cvt_grid = (int)(WE / 256);
  cvt_f32_bf16_kernel<<<cvt_grid, 256, 0, stream>>>(Wq, Wqb, (int)WE);
  cvt_f32_bf16_kernel<<<cvt_grid, 256, 0, stream>>>(Wk, Wkb, (int)WE);
  cvt_f32_bf16_kernel<<<cvt_grid, 256, 0, stream>>>(Wv, Wvb, (int)WE);
  cvt_f32_bf16_kernel<<<cvt_grid, 256, 0, stream>>>(Wo, Wob, (int)WE);

  // 2) Q/K/V projections (fp32 in, bf16 out in attention-friendly layouts)
  dim3 gg(DMODEL / 64, (BATCH * SEQ) / 128, 1);
  gemm1024_kernel<0, 0><<<gg, 256, 0, stream>>>(query, Wqb, bq, Qbf);
  gemm1024_kernel<0, 0><<<gg, 256, 0, stream>>>(key,   Wkb, bk, Kbf);
  gemm1024_kernel<0, 1><<<gg, 256, 0, stream>>>(value, Wvb, bv, Vtb);

  // 3) flash attention (async double-buffered K/V staging)
  attn_kernel<<<dim3(SEQ / 128, NHEAD, BATCH), 256, 0, stream>>>(Qbf, Kbf, Vtb, Obf);

  // 4) output projection (bf16 in via async-to-LDS, fp32 out + bias)
  gemm1024_kernel<1, 2><<<gg, 256, 0, stream>>>(Obf, Wob, bo, d_out);
}
